// EMAVectorQuantizer_88021059764278
// MI455X (gfx1250) — compile-verified, matrix operand order checked by reading, not run on hardware
//
#include <hip/hip_runtime.h>
#include <hip/hip_bf16.h>

#define D 256
#define K_EMB 8192
#define N_VEC 32768
#define CHUNK 64
#define ROWSP 264   // padded LDS row stride in bf16 elements (528 B -> bank stride 4)
#define ROWS_PER_BLOCK 256   // 8 waves x 32 rows
#define NBLOCKS (N_VEC / ROWS_PER_BLOCK)

typedef __attribute__((ext_vector_type(16))) __bf16 v16bf;
typedef __attribute__((ext_vector_type(8)))  __bf16 v8bf;
typedef __attribute__((ext_vector_type(8)))  float  v8f;

// ---------------- prep: embedding f32 -> bf16, half squared norms ----------------
__global__ void vq_prep(const float* __restrict__ emb, __bf16* __restrict__ emb_bf,
                        float* __restrict__ halfnorm) {
  const int row = blockIdx.x, t = threadIdx.x;
  const float e = emb[(size_t)row * D + t];
  emb_bf[(size_t)row * D + t] = (__bf16)e;
  __shared__ float red[256];
  red[t] = e * e;
  __syncthreads();
  for (int s = 128; s > 0; s >>= 1) {
    if (t < s) red[t] += red[t + s];
    __syncthreads();
  }
  if (t == 0) halfnorm[row] = 0.5f * red[0];
}

// load one 16x32 bf16 A-fragment (ISA 16-bit A layout) from an f32 row
__device__ __forceinline__ v16bf load_a_frag(const float* __restrict__ xr, int ks, int g) {
  const int k0 = ks * 32 + g * 8;
  float4 f0 = *(const float4*)(xr + k0);
  float4 f1 = *(const float4*)(xr + k0 + 4);
  float4 f2 = *(const float4*)(xr + k0 + 16);
  float4 f3 = *(const float4*)(xr + k0 + 20);
  v16bf t;
  t[0] = (__bf16)f0.x;  t[1] = (__bf16)f0.y;  t[2] = (__bf16)f0.z;  t[3] = (__bf16)f0.w;
  t[4] = (__bf16)f1.x;  t[5] = (__bf16)f1.y;  t[6] = (__bf16)f1.z;  t[7] = (__bf16)f1.w;
  t[8] = (__bf16)f2.x;  t[9] = (__bf16)f2.y;  t[10] = (__bf16)f2.z; t[11] = (__bf16)f2.w;
  t[12] = (__bf16)f3.x; t[13] = (__bf16)f3.y; t[14] = (__bf16)f3.z; t[15] = (__bf16)f3.w;
  return t;
}

// ---------------- fused distances + argmin + gather + loss partials ----------------
__global__ void __launch_bounds__(256) vq_main(
    const float* __restrict__ x, const float* __restrict__ emb,
    const __bf16* __restrict__ emb_bf, const float* __restrict__ halfnorm,
    float* __restrict__ out_q, int* __restrict__ idx_out,
    float* __restrict__ partial) {
  __shared__ __align__(16) __bf16 smem[CHUNK * ROWSP];
  __shared__ float wloss[8];

  const int tid  = threadIdx.x;
  const int wave = tid >> 5, lane = tid & 31;
  const int g    = lane >> 4, lrow = lane & 15;       // half-wave group, row/col within tile
  const int rBase = blockIdx.x * ROWS_PER_BLOCK + wave * 32; // this wave's 32 input rows

  // ---- two resident A-fragment sets: rows [rBase, rBase+16) and [rBase+16, rBase+32) ----
  v16bf a0[8], a1[8];
  {
    const float* xr0 = x + (size_t)(rBase + lrow) * D;
    const float* xr1 = x + (size_t)(rBase + 16 + lrow) * D;
#pragma unroll
    for (int ks = 0; ks < 8; ++ks) {
      a0[ks] = load_a_frag(xr0, ks, g);
      a1[ks] = load_a_frag(xr1, ks, g);
    }
  }

  float best0[8], best1[8];
  int   bidx0[8], bidx1[8];
#pragma unroll
  for (int v = 0; v < 8; ++v) {
    best0[v] = -3.4e38f; bidx0[v] = 0;
    best1[v] = -3.4e38f; bidx1[v] = 0;
  }

  const int srow = tid & 63, sq = tid >> 6; // staging assignment: row, quarter

  for (int chunk = 0; chunk < K_EMB / CHUNK; ++chunk) {
    const int eBase = chunk * CHUNK;
    // ---- stage CHUNK bf16 embedding rows into padded LDS ----
    {
      const v8bf* src = (const v8bf*)(emb_bf + (size_t)(eBase + srow) * D + sq * 64);
      __bf16* dst = smem + srow * ROWSP + sq * 64;
#pragma unroll
      for (int i = 0; i < 8; ++i) *(v8bf*)(dst + i * 8) = src[i];
      if (eBase + CHUNK < K_EMB)
        __builtin_prefetch((const void*)(emb_bf + (size_t)(eBase + CHUNK + srow) * D + sq * 64), 0, 1);
    }
    __syncthreads();

#pragma unroll 1   // keep only one column group in flight -> bounded VGPR pressure
    for (int cg = 0; cg < 4; ++cg) {
      const int colBase = eBase + cg * 16;
      const float hn = halfnorm[colBase + lrow]; // this lane's column norm
      v8f c0 = {0.f, 0.f, 0.f, 0.f, 0.f, 0.f, 0.f, 0.f};
      v8f c1 = {0.f, 0.f, 0.f, 0.f, 0.f, 0.f, 0.f, 0.f};
#pragma unroll
      for (int ks = 0; ks < 8; ++ks) {
        const __bf16* bp = smem + (cg * 16 + lrow) * ROWSP + ks * 32 + g * 8;
        v8bf b0 = *(const v8bf*)bp;
        v8bf b1 = *(const v8bf*)(bp + 16);
        v16bf b = __builtin_shufflevector(b0, b1, 0, 1, 2, 3, 4, 5, 6, 7,
                                                  8, 9, 10, 11, 12, 13, 14, 15);
        // one B fragment feeds two WMMAs (two M tiles) -> half the LDS bytes per FLOP
        c0 = __builtin_amdgcn_wmma_f32_16x16x32_bf16(false, a0[ks], false, b,
                                                     (short)0, c0, false, false);
        c1 = __builtin_amdgcn_wmma_f32_16x16x32_bf16(false, a1[ks], false, b,
                                                     (short)0, c1, false, false);
      }
      // C layout: lane holds column n = lrow; VGPR v holds row 8*g + v
#pragma unroll
      for (int v = 0; v < 8; ++v) {
        const float s0 = c0[v] - hn; // maximize x.e - 0.5||e||^2 == argmin distance
        if (s0 > best0[v]) { best0[v] = s0; bidx0[v] = colBase + lrow; }
        const float s1 = c1[v] - hn;
        if (s1 > best1[v]) { best1[v] = s1; bidx1[v] = colBase + lrow; }
      }
    }
    __syncthreads();
  }

  // ---- argmax reduction across the 16 lanes of each half-wave (both tiles) ----
#pragma unroll
  for (int v = 0; v < 8; ++v) {
    float s0 = best0[v]; int i0 = bidx0[v];
    float s1 = best1[v]; int i1 = bidx1[v];
#pragma unroll
    for (int off = 1; off < 16; off <<= 1) {
      float so0 = __shfl_xor(s0, off, 16);
      int   io0 = __shfl_xor(i0, off, 16);
      if (so0 > s0 || (so0 == s0 && io0 < i0)) { s0 = so0; i0 = io0; }
      float so1 = __shfl_xor(s1, off, 16);
      int   io1 = __shfl_xor(i1, off, 16);
      if (so1 > s1 || (so1 == s1 && io1 < i1)) { s1 = so1; i1 = io1; }
    }
    bidx0[v] = i0;
    bidx1[v] = i1;
  }

  // ---- gather winners (f32), write quantized output + indices, loss partials ----
  // fully unrolled: best/bidx always indexed by compile-time constants (no scratch)
  float lacc = 0.f;
#pragma unroll
  for (int r = 0; r < 32; ++r) {
    const int v = r & 7;
    const int srcLane = ((r >> 3) & 1) * 16; // rows 8g+v live in lane-group g
    const int id = (r < 16) ? __shfl(bidx0[v], srcLane, 32)
                            : __shfl(bidx1[v], srcLane, 32);
    const int gr = rBase + r;
    if (lane == 0) idx_out[gr] = id;
    const float* er = emb + (size_t)id * D + lane * 8;
    const float* xr = x + (size_t)gr * D + lane * 8;
    float* qr = out_q + (size_t)gr * D + lane * 8;
    float4 e0 = *(const float4*)er, e1 = *(const float4*)(er + 4);
    float4 x0 = *(const float4*)xr, x1 = *(const float4*)(xr + 4);
    *(float4*)qr = e0;
    *(float4*)(qr + 4) = e1;
    float d0 = e0.x - x0.x, d1 = e0.y - x0.y, d2 = e0.z - x0.z, d3 = e0.w - x0.w;
    float d4 = e1.x - x1.x, d5 = e1.y - x1.y, d6 = e1.z - x1.z, d7 = e1.w - x1.w;
    lacc += d0 * d0 + d1 * d1 + d2 * d2 + d3 * d3 + d4 * d4 + d5 * d5 + d6 * d6 + d7 * d7;
  }
#pragma unroll
  for (int off = 16; off > 0; off >>= 1) lacc += __shfl_xor(lacc, off, 32);
  if (lane == 0) wloss[wave] = lacc;
  __syncthreads();
  if (tid == 0) {
    float s = 0.f;
    for (int w = 0; w < 8; ++w) s += wloss[w];
    partial[blockIdx.x] = s;
  }
}

// ---------------- deterministic loss finalize ----------------
__global__ void vq_finalize(const float* __restrict__ partial, float* __restrict__ loss_out) {
  __shared__ float red[256];
  const int t = threadIdx.x;
  red[t] = (t < NBLOCKS) ? partial[t] : 0.f;
  __syncthreads();
  for (int s = 128; s > 0; s >>= 1) {
    if (t < s) red[t] += red[t + s];
    __syncthreads();
  }
  if (t == 0) loss_out[0] = red[0] * (0.25f / (float)((size_t)N_VEC * D));
}

extern "C" void kernel_launch(void* const* d_in, const int* in_sizes, int n_in,
                              void* d_out, int out_size, void* d_ws, size_t ws_size,
                              hipStream_t stream) {
  (void)in_sizes; (void)n_in; (void)out_size; (void)ws_size;
  const float* x   = (const float*)d_in[0]; // [8,4096,256] f32
  const float* emb = (const float*)d_in[1]; // [8192,256]   f32

  // workspace: bf16 embedding (4 MB) | half-norms (32 KB) | block loss partials
  __bf16* emb_bf   = (__bf16*)d_ws;
  float*  halfnorm = (float*)((char*)d_ws + (size_t)K_EMB * D * 2);
  float*  partial  = halfnorm + K_EMB;

  // d_out: quantized_st [N,D] f32 | loss scalar | indices [N] (int32 bit pattern)
  float* out_q = (float*)d_out;
  float* loss  = out_q + (size_t)N_VEC * D;
  int*   idx   = (int*)d_out + ((size_t)N_VEC * D + 1);

  vq_prep<<<K_EMB, 256, 0, stream>>>(emb, emb_bf, halfnorm);
  vq_main<<<NBLOCKS, 256, 0, stream>>>(x, emb, emb_bf, halfnorm, out_q, idx, partial);
  vq_finalize<<<1, 256, 0, stream>>>(partial, loss);
}
